// BodyOrderExpansionBlock_40046275067963
// MI455X (gfx1250) — matrix operand order repaired and checked by reading, not compiled, not run on hardware
//
#include <hip/hip_runtime.h>
#include <hip/hip_bf16.h>

typedef __bf16 bf16_t;
typedef __bf16 v8bf  __attribute__((ext_vector_type(8)));
typedef __bf16 v16bf __attribute__((ext_vector_type(16)));
typedef float  v8f   __attribute__((ext_vector_type(8)));

#define NN 24576
#define FF 512
#define KK3 1536

// ----------------------------- conversion kernels -----------------------------
__global__ __launch_bounds__(256) void cvt_f32_to_bf16(const float* __restrict__ s,
                                                       bf16_t* __restrict__ d, int n) {
  int i = blockIdx.x * 256 + threadIdx.x;
  if (i < n) d[i] = (bf16_t)s[i];
}

// dst[(C x R) row-major] = transpose of src[(R x C) row-major], f32 -> bf16
__global__ __launch_bounds__(256) void transpose_f32_to_bf16(const float* __restrict__ s,
                                                             bf16_t* __restrict__ d,
                                                             int R, int C) {
  int i = blockIdx.x * 256 + threadIdx.x;
  if (i < R * C) {
    int r = i / C, q = i - r * C;
    d[(size_t)q * R + r] = (bf16_t)s[(size_t)r * C + q];
  }
}

// ----------------------------- WMMA fragment loaders -----------------------------
// LDS tiles are row-major [16 rows][LD=40 bf16], 80B pitch (16B aligned).
// A operand, 16-bit 16x32 (ISA 7.12.2): lane = M mod 16; lanes 0-15 hold
// K = {0..7, 16..23}, lanes 16-31 hold K = {8..15, 24..31}.
__device__ __forceinline__ v16bf load_fragA(const bf16_t* __restrict__ p0, int lane) {
  const int r  = lane & 15;
  const int kb = (lane >> 4) << 3;  // 0 or 8
  const bf16_t* p = p0 + r * 40;
  v8bf lo = *(const v8bf*)(p + kb);        // K = kb .. kb+7
  v8bf hi = *(const v8bf*)(p + 16 + kb);   // K = 16+kb .. 16+kb+7
  return __builtin_shufflevector(lo, hi, 0,1,2,3,4,5,6,7,8,9,10,11,12,13,14,15);
}
// B operand, 16-bit 32x16: lane = N mod 16; lanes 0-15 hold
// K = {0..7, 16..23}, lanes 16-31 hold K = {8..15, 24..31} (same as A).
__device__ __forceinline__ v16bf load_fragB(const bf16_t* __restrict__ p0, int lane) {
  const int r  = lane & 15;
  const int kh = (lane >> 4) << 3;  // 0 or 8
  const bf16_t* p = p0 + r * 40 + kh;
  v8bf lo = *(const v8bf*)p;
  v8bf hi = *(const v8bf*)(p + 16);
  return __builtin_shufflevector(lo, hi, 0,1,2,3,4,5,6,7,8,9,10,11,12,13,14,15);
}

// ----------------------------- tiled WMMA GEMM -----------------------------
// C(M x 512) = A(M x 512 bf16) @ Bt^T, Bt is (512 x K) bf16, K in {512, 1536}.
// MODE 0: out = bf16(C + bias)                                (h = x@W1 + b1)
// MODE 1: out = bf16(C)                                       (g = h@W2)
// MODE 2: A column k reads u[n, k&511]; per 512-wide K-segment l the partial
//         product is folded with per-row scale c[n,l]; out = f32(silu(C+bias)).
template <int MODE>
__global__ __launch_bounds__(256) void wmma_gemm(const bf16_t* __restrict__ A,
                                                 const bf16_t* __restrict__ Bt,
                                                 const float*  __restrict__ bias,
                                                 const float*  __restrict__ cscale,
                                                 bf16_t* __restrict__ outBf,
                                                 float*  __restrict__ outF,
                                                 int K) {
  constexpr int LD = 40;                       // padded row pitch (bf16 elems)
  __shared__ bf16_t sA[2][128 * LD];
  __shared__ bf16_t sB[2][128 * LD];

  const int tid   = threadIdx.x;
  const int lane  = tid & 31;
  const int wave  = tid >> 5;
  const int waveM = wave & 1;                  // 2 waves along M (64 rows each)
  const int waveN = wave >> 1;                 // 4 waves along N (32 cols each)
  const int rowBase = blockIdx.y * 128;
  const int colBase = blockIdx.x * 128;
  const int halfSel = (lane >> 4) << 3;        // C/D: M = r + halfSel
  const int nlane   = lane & 15;               // C/D: N = lane mod 16

  // staging map: 256 threads = 128 rows x 2 halves of 16 bf16 (32B) each
  const int srow  = tid >> 1;
  const int shalf = tid & 1;
  const size_t aRow = (size_t)(rowBase + srow) * 512;
  const size_t bRow = (size_t)(colBase + srow) * (size_t)K;
  const int KT = K >> 5;

  v8f acc[4][2];                               // running (segment) accumulator
  v8f accF[4][2];                              // folded result (MODE 2 only)
  {
    v8f z = {};
    #pragma unroll
    for (int m = 0; m < 4; ++m)
      #pragma unroll
      for (int n = 0; n < 2; ++n) { acc[m][n] = z; accF[m][n] = z; }
  }

  // async global->LDS staging (CDNA5): LDS dest addr = AS(3) offset = low 32
  // bits of the generic pointer; INST_OFFSET applies to both addresses.
  auto issueTile = [&](int kt, int buf) {
    const int k0 = kt << 5;
    const int ks = k0 + shalf * 16;
    const int acol = (MODE == 2) ? (ks & 511) : ks;
    unsigned long long ga = (unsigned long long)(const void*)(A + aRow + acol);
    unsigned long long gb = (unsigned long long)(const void*)(Bt + bRow + ks);
    unsigned la = (unsigned)(unsigned long long)(const void*)&sA[buf][srow * LD + shalf * 16];
    unsigned lb = (unsigned)(unsigned long long)(const void*)&sB[buf][srow * LD + shalf * 16];
    asm volatile(
        "global_load_async_to_lds_b128 %0, %2, off\n\t"
        "global_load_async_to_lds_b128 %0, %2, off offset:16\n\t"
        "global_load_async_to_lds_b128 %1, %3, off\n\t"
        "global_load_async_to_lds_b128 %1, %3, off offset:16"
        :: "v"(la), "v"(lb), "v"(ga), "v"(gb) : "memory");
  };

  issueTile(0, 0);
  asm volatile("s_wait_asynccnt 0" ::: "memory");
  __syncthreads();

  for (int kt = 0; kt < KT; ++kt) {
    const int cur = kt & 1;
    if (kt + 1 < KT) {
      issueTile(kt + 1, cur ^ 1);              // async copy overlaps the WMMAs
      if (kt + 2 < KT) {                       // L2 prefetch two tiles ahead
        const int kp = (kt + 2) << 5;
        __builtin_prefetch(A + aRow + ((MODE == 2) ? (kp & 511) : kp), 0, 0);
        __builtin_prefetch(Bt + bRow + kp, 0, 0);
      }
    }

    v16bf af[4], bfg[2];
    #pragma unroll
    for (int m = 0; m < 4; ++m)
      af[m] = load_fragA(&sA[cur][(waveM * 64 + m * 16) * LD], lane);
    #pragma unroll
    for (int n = 0; n < 2; ++n)
      bfg[n] = load_fragB(&sB[cur][(waveN * 32 + n * 16) * LD], lane);

    #pragma unroll
    for (int m = 0; m < 4; ++m)
      #pragma unroll
      for (int n = 0; n < 2; ++n)
        acc[m][n] = __builtin_amdgcn_wmma_f32_16x16x32_bf16(
            false, af[m], false, bfg[n], (short)0, acc[m][n], false, false);

    if constexpr (MODE == 2) {                 // fold at 512-wide segment ends
      if (((kt + 1) & 15) == 0) {
        const int l = kt >> 4;
        #pragma unroll
        for (int m = 0; m < 4; ++m) {
          #pragma unroll
          for (int r = 0; r < 8; ++r) {
            const int row = rowBase + waveM * 64 + m * 16 + r + halfSel;
            const float cr = cscale[(size_t)row * 3 + l];
            accF[m][0][r] += cr * acc[m][0][r];
            accF[m][1][r] += cr * acc[m][1][r];
            acc[m][0][r] = 0.f;
            acc[m][1][r] = 0.f;
          }
        }
      }
    }

    asm volatile("s_wait_asynccnt 0" ::: "memory");
    __syncthreads();
  }

  // epilogue
  #pragma unroll
  for (int n = 0; n < 2; ++n) {
    const int col = colBase + waveN * 32 + n * 16 + nlane;
    float bv = 0.f;
    if constexpr (MODE != 1) bv = bias[col];
    #pragma unroll
    for (int m = 0; m < 4; ++m) {
      #pragma unroll
      for (int r = 0; r < 8; ++r) {
        const int row = rowBase + waveM * 64 + m * 16 + r + halfSel;
        float v = ((MODE == 2) ? accF[m][n][r] : acc[m][n][r]) + bv;
        if constexpr (MODE == 2) {
          v = v * (1.f / (1.f + __expf(-v)));  // SiLU
          outF[(size_t)row * 512 + col] = v;
        } else {
          outBf[(size_t)row * 512 + col] = (bf16_t)v;
        }
      }
    }
  }
}

// ----------------------------- per-row epilogue ops -----------------------------
// s[n] = g[n,:]·W3 ; u = bf16(g^2) ; c[n,l] = segment sums of chi^2 ; chi_bo = chi*s
__global__ __launch_bounds__(256) void rowops(const bf16_t* __restrict__ g,
                                              const float*  __restrict__ W3,
                                              const float*  __restrict__ chi,
                                              bf16_t* __restrict__ u,
                                              float*  __restrict__ c,
                                              float*  __restrict__ chi_bo) {
  const int wave = threadIdx.x >> 5;
  const int lane = threadIdx.x & 31;
  const int n = blockIdx.x * 8 + wave;
  const bf16_t* gr = g + (size_t)n * 512;
  bf16_t* ur = u + (size_t)n * 512;
  float s = 0.f;
  #pragma unroll
  for (int t = 0; t < 16; ++t) {
    const int j = lane + t * 32;
    const float gv = (float)gr[j];
    s += gv * W3[j];
    ur[j] = (bf16_t)(gv * gv);
  }
  #pragma unroll
  for (int off = 16; off; off >>= 1) s += __shfl_xor(s, off, 32);
  if (lane == 0) {
    const float* cr = chi + (size_t)n * 15;
    float* cb = chi_bo + (size_t)n * 15;
    float c0 = 0.f, c1 = 0.f, c2 = 0.f;
    #pragma unroll
    for (int m = 0; m < 15; ++m) {
      const float v = cr[m];
      cb[m] = v * s;
      const float v2 = v * v;
      if (m < 3) c0 += v2; else if (m < 8) c1 += v2; else c2 += v2;
    }
    c[(size_t)n * 3 + 0] = c0;
    c[(size_t)n * 3 + 1] = c1;
    c[(size_t)n * 3 + 2] = c2;
  }
}

// ----------------------------- launcher -----------------------------
extern "C" void kernel_launch(void* const* d_in, const int* in_sizes, int n_in,
                              void* d_out, int out_size, void* d_ws, size_t ws_size,
                              hipStream_t stream) {
  const float* x   = (const float*)d_in[0];   // (NN, 512)
  const float* chi = (const float*)d_in[1];   // (NN, 15)
  // d_in[2] = z_one_hot: unused by the reference
  const float* W1  = (const float*)d_in[3];   // (512, 512)
  const float* b1  = (const float*)d_in[4];   // (512,)
  const float* W2  = (const float*)d_in[5];   // (512, 512)
  const float* W3  = (const float*)d_in[6];   // (512, 1)
  const float* W4  = (const float*)d_in[7];   // (1536, 512)
  const float* b4  = (const float*)d_in[8];   // (512,)

  float* outF   = (float*)d_out;              // x_bo: NN*512 f32
  float* chi_bo = outF + (size_t)NN * 512;    // chi_bo: NN*15 f32

  // workspace carve-up (slot reuse: x_bf -> g_bf, h_bf -> u_bf)
  char* ws = (char*)d_ws;
  const size_t S1 = (size_t)NN * 512 * sizeof(bf16_t);   // 25 MB
  bf16_t* slot0 = (bf16_t*)ws;                 // x_bf, then g_bf
  bf16_t* slot1 = (bf16_t*)(ws + S1);          // h_bf, then u_bf
  bf16_t* W1t   = (bf16_t*)(ws + 2 * S1);                      // 512x512
  bf16_t* W2t   = W1t + (size_t)512 * 512;                     // 512x512
  bf16_t* W4t   = W2t + (size_t)512 * 512;                     // 512x1536
  float*  cseg  = (float*)(W4t + (size_t)512 * 1536);          // NN x 3

  // 1) precision conversion / weight transposition (weights stay resident in L2)
  cvt_f32_to_bf16<<<(NN * 512 + 255) / 256, 256, 0, stream>>>(x, slot0, NN * 512);
  transpose_f32_to_bf16<<<(512 * 512 + 255) / 256, 256, 0, stream>>>(W1, W1t, 512, 512);
  transpose_f32_to_bf16<<<(512 * 512 + 255) / 256, 256, 0, stream>>>(W2, W2t, 512, 512);
  transpose_f32_to_bf16<<<(1536 * 512 + 255) / 256, 256, 0, stream>>>(W4, W4t, 1536, 512);

  const dim3 grid(512 / 128, NN / 128);        // (4, 192)

  // 2) h = bf16(x @ W1 + b1)
  wmma_gemm<0><<<grid, 256, 0, stream>>>(slot0, W1t, b1, nullptr, slot1, nullptr, 512);
  // 3) g = bf16(h @ W2)   (overwrites x_bf slot)
  wmma_gemm<1><<<grid, 256, 0, stream>>>(slot1, W2t, nullptr, nullptr, slot0, nullptr, 512);
  // 4) s = g·W3, u = g^2, c = seg-sums(chi^2), chi_bo = chi*s  (u overwrites h slot)
  rowops<<<NN / 8, 256, 0, stream>>>(slot0, W3, chi, slot1, cseg, chi_bo);
  // 5) x_bo = silu( [c0*u | c1*u | c2*u] @ W4 + b4 )
  wmma_gemm<2><<<grid, 256, 0, stream>>>(slot1, W4t, b4, cseg, nullptr, outF, KK3);
}
